// EncoderRNN_9440338117094
// MI455X (gfx1250) — matrix operand order repaired and checked by reading, not compiled, not run on hardware
//
#include <hip/hip_runtime.h>
#include <hip/hip_bf16.h>
#include <math.h>
#include <cstdint>

typedef __attribute__((ext_vector_type(16))) __bf16 v16bf;
typedef __attribute__((ext_vector_type(8)))  __bf16 v8bf;
typedef __attribute__((ext_vector_type(4)))  __bf16 v4bf;
typedef __attribute__((ext_vector_type(8)))  float  v8f;
typedef __attribute__((ext_vector_type(4)))  float  v4f;
typedef __attribute__((ext_vector_type(4)))  int    v4i;

#define B_    64
#define T_    2048
#define E_    256
#define H_    256
#define G_    768        // 3*H (r,z,n gate blocks)
#define LDA_  264        // padded bf16 LDS row stride (16B pad -> bank rotation)
#define LDGH_ 772        // padded f32 LDS row stride for gate pre-activations
#define LDGI_ 776        // padded bf16 LDS row stride for gi tiles / staging

#if __has_builtin(__builtin_amdgcn_global_load_async_to_lds_b128) && \
    __has_builtin(__builtin_amdgcn_s_wait_asynccnt)
#define USE_ASYNC_LDS 1
#else
#define USE_ASYNC_LDS 0
#endif

#if __has_builtin(__builtin_amdgcn_sched_barrier)
#define SCHED_FENCE() __builtin_amdgcn_sched_barrier(0)
#else
#define SCHED_FENCE()
#endif

#if USE_ASYNC_LDS
// Builtin prototype (from clang diagnostic): arg0 = int4 addrspace(1)*,
// arg1 = int4 addrspace(3)*, then two constant ints (offset, cpol).
typedef __attribute__((address_space(1))) v4i* gbl4p;
typedef __attribute__((address_space(3))) v4i* lds4p;

// Per-lane 16B async copy global -> LDS (tracked by ASYNCcnt). AS casts go
// through integers: generic LDS addresses carry the LDS offset in the low 32
// bits (flat-LDS mapping truncates); global VAs pass through unchanged.
__device__ __forceinline__ void async_copy16(const __bf16* g, __bf16* l) {
  __builtin_amdgcn_global_load_async_to_lds_b128(
      (gbl4p)(uintptr_t)g, (lds4p)(uint32_t)(uintptr_t)l, 0, 0);
}
#endif

// Build a 16x32 bf16 A-fragment for v_wmma_f32_16x16x32_bf16 from a row-major
// bf16 tile in LDS. Lane L (row M=L&15) holds K = kb..kb+15,
// kb = (lane<16 ? 0 : 16) within the 32-wide K tile.
__device__ __forceinline__ v16bf load_a_frag(const __bf16* smem, int lane, int kt) {
  const int row = lane & 15;
  const int kb  = kt * 32 + ((lane >> 4) << 4);
  v8bf lo = *(const v8bf*)(smem + row * LDA_ + kb);
  v8bf hi = *(const v8bf*)(smem + row * LDA_ + kb + 8);
  v16bf a;
#pragma unroll
  for (int e = 0; e < 8; ++e) { a[e] = lo[e]; a[e + 8] = hi[e]; }
  return a;
}

// ---------------- Phase 0: weights f32 -> bf16 ------------------------------
__global__ void k_cvt_w(const float* __restrict__ wih, const float* __restrict__ whh,
                        __bf16* __restrict__ wihb, __bf16* __restrict__ whhb) {
  int i = blockIdx.x * blockDim.x + threadIdx.x;
  if (i < G_ * E_) {
    wihb[i] = (__bf16)wih[i];
    whhb[i] = (__bf16)whh[i];
  }
}

// ---------------- Phase 1: gi[t][b][g] = emb[seq] @ w_ih^T + b_ih -----------
// 1024 blocks x 768 threads (24 waves). Block computes 128 rows (8 M-tiles).
// Each wave owns 32 gate columns; its w_ih B-fragments stay in VGPRs for all
// 8 M-tiles. Passes are fenced so only one accumulator chain is live; the one
// residual 32B reload (160-VGPR budget at 24 waves) is off the critical path.
__global__ void __launch_bounds__(768, 1)
k_gi(const int* __restrict__ seq, const float* __restrict__ emb,
     const __bf16* __restrict__ wihb, const float* __restrict__ b_ih,
     __bf16* __restrict__ gi) {
  __shared__ __bf16 xbf[16 * LDA_];
  __shared__ __bf16 ost[16 * LDGI_];
  const int tid  = threadIdx.x;
  const int lane = tid & 31;
  const int wave = tid >> 5;

  v16bf Bf[2][8];
  float bias[2];
#pragma unroll
  for (int j = 0; j < 2; ++j) {
    const int g = wave * 32 + j * 16 + (lane & 15);
    bias[j] = b_ih[g];
#pragma unroll
    for (int kt = 0; kt < 8; ++kt)
      Bf[j][kt] = *(const v16bf*)(wihb + g * E_ + kt * 32 + ((lane >> 4) << 4));
  }

  const int mbase = blockIdx.x * 128;
  for (int tile = 0; tile < 8; ++tile) {
    const int m0 = mbase + tile * 16;
    // gather 16 embedding rows, 4-wide, cvt f32->bf16 into LDS
    for (int v = tid; v < 1024; v += 768) {
      const int r = v >> 6, c = (v & 63) << 2;
      const int m = m0 + r;
      const int b = m & 63, t = m >> 6;            // row m = t*64 + b
      const int tok = seq[b * T_ + t];
      v4f x4 = *(const v4f*)(emb + (size_t)tok * E_ + c);
      v4bf xb;
#pragma unroll
      for (int e = 0; e < 4; ++e) xb[e] = (__bf16)x4[e];
      *(v4bf*)(xbf + r * LDA_ + c) = xb;
    }
    __syncthreads();

#pragma unroll
    for (int j = 0; j < 2; ++j) {                   // one acc live at a time
      v8f acc = {0.f,0.f,0.f,0.f,0.f,0.f,0.f,0.f};
#pragma unroll
      for (int kt = 0; kt < 8; ++kt) {
        v16bf a = load_a_frag(xbf, lane, kt);
        acc = __builtin_amdgcn_wmma_f32_16x16x32_bf16(false, a, false, Bf[j][kt],
                                                      (short)0, acc, false, false);
      }
      const int g = wave * 32 + j * 16 + (lane & 15);
#pragma unroll
      for (int i = 0; i < 8; ++i) {
        const int r = i + ((lane >> 4) << 3);
        ost[r * LDGI_ + g] = (__bf16)(acc[i] + bias[j]);
      }
      SCHED_FENCE();   // end acc's live range before the next pass begins
    }
    __syncthreads();

    // coalesced b128 stores: 12288 bf16 = 1536 chunks of 8
    for (int ch = tid; ch < 1536; ch += 768) {
      const int r = ch / 96, c8 = (ch % 96) * 8;
      *(v8bf*)(gi + (size_t)(m0 + r) * G_ + c8) = *(const v8bf*)(ost + r * LDGI_ + c8);
    }
    // next iteration's gather (xbf) and these ost reads don't alias; the
    // gather barrier orders ost reuse.
  }
}

// ---------------- Phase 2: sequential GRU scan ------------------------------
// 4 blocks x 512 threads (16 waves, 4/SIMD -> 256-VGPR budget). Block owns 16
// batch rows for all T steps. Each wave owns 48 gate columns: w_hh bf16 lives
// entirely in VGPRs (Bf[3][8] = 192 regs), 3 accumulator chains share each
// A-fragment load. gi tiles are double-buffered in LDS via async global->LDS
// copies issued one step ahead (ASYNCcnt).
__global__ void __launch_bounds__(512, 1)
k_scan(const __bf16* __restrict__ whhb, const float* __restrict__ b_hh,
       const __bf16* __restrict__ gi, float* __restrict__ out) {
  __shared__ float  hs [16 * H_];     // master f32 hidden state
  __shared__ __bf16 hbf[16 * LDA_];   // bf16 copy for WMMA A-fragments
  __shared__ float  gh [16 * LDGH_];  // h @ w_hh^T + b_hh
#if USE_ASYNC_LDS
  __shared__ __bf16 gis[2][16 * LDGI_];
#endif
  const int tid  = threadIdx.x;
  const int lane = tid & 31;
  const int wave = tid >> 5;          // 0..15
  const int bbase = blockIdx.x * 16;

  v16bf Bf[3][8];
  float bias[3];
#pragma unroll
  for (int j = 0; j < 3; ++j) {
    const int g = wave * 48 + j * 16 + (lane & 15);
    bias[j] = b_hh[g];
#pragma unroll
    for (int kt = 0; kt < 8; ++kt)
      Bf[j][kt] = *(const v16bf*)(whhb + g * H_ + kt * 32 + ((lane >> 4) << 4));
  }

  for (int idx = tid; idx < 16 * H_; idx += 512) {   // h0 = 0
    hs[idx] = 0.f;
    hbf[(idx >> 8) * LDA_ + (idx & 255)] = (__bf16)0.f;
  }

#if USE_ASYNC_LDS
  // loop-invariant async chunk coordinates (3 x 16B chunks per thread)
  int rc[3], cc[3];
#pragma unroll
  for (int i = 0; i < 3; ++i) {
    const int ch = tid + i * 512;                    // 0..1535
    rc[i] = ch / 96;
    cc[i] = (ch % 96) * 8;
  }
  {                                                  // preload gi tile t=0
    const __bf16* g0 = gi + (size_t)bbase * G_;
#pragma unroll
    for (int i = 0; i < 3; ++i)
      async_copy16(g0 + rc[i] * G_ + cc[i], &gis[0][rc[i] * LDGI_ + cc[i]]);
  }
#endif
  // loop-invariant gate-phase coordinates (vec4 chunks: tid, tid+512)
  const int rgA = tid >> 6,         cgA = (tid & 63) << 2;   // rows 0..7
  const int rgB = (tid + 512) >> 6, cgB = cgA;               // rows 8..15
  __syncthreads();

  for (int t = 0; t < T_; ++t) {
#if USE_ASYNC_LDS
    if (t + 1 < T_) {                                // stage gi[t+1]
      const __bf16* gn = gi + ((size_t)(t + 1) * B_ + bbase) * G_;
      __bf16* buf = &gis[(t + 1) & 1][0];
#pragma unroll
      for (int i = 0; i < 3; ++i)
        async_copy16(gn + rc[i] * G_ + cc[i], buf + rc[i] * LDGI_ + cc[i]);
    }
#endif
    // gh = h @ w_hh^T + b_hh : 3 interleaved chains sharing each A fragment
    v8f acc0 = {0.f,0.f,0.f,0.f,0.f,0.f,0.f,0.f};
    v8f acc1 = {0.f,0.f,0.f,0.f,0.f,0.f,0.f,0.f};
    v8f acc2 = {0.f,0.f,0.f,0.f,0.f,0.f,0.f,0.f};
#pragma unroll
    for (int kt = 0; kt < 8; ++kt) {
      v16bf a = load_a_frag(hbf, lane, kt);
      acc0 = __builtin_amdgcn_wmma_f32_16x16x32_bf16(false, a, false, Bf[0][kt],
                                                     (short)0, acc0, false, false);
      acc1 = __builtin_amdgcn_wmma_f32_16x16x32_bf16(false, a, false, Bf[1][kt],
                                                     (short)0, acc1, false, false);
      acc2 = __builtin_amdgcn_wmma_f32_16x16x32_bf16(false, a, false, Bf[2][kt],
                                                     (short)0, acc2, false, false);
    }
#pragma unroll
    for (int j = 0; j < 3; ++j) {
      const v8f acc = (j == 0) ? acc0 : (j == 1) ? acc1 : acc2;
      const int g = wave * 48 + j * 16 + (lane & 15);
#pragma unroll
      for (int i = 0; i < 8; ++i) {
        const int r = i + ((lane >> 4) << 3);
        gh[r * LDGH_ + g] = acc[i] + bias[j];
      }
    }
#if USE_ASYNC_LDS
    if (t + 1 < T_) __builtin_amdgcn_s_wait_asynccnt(3);   // gi[t] landed
    else            __builtin_amdgcn_s_wait_asynccnt(0);
#endif
    __syncthreads();

    // gate math, 4-wide: chunk A (rows 0..7) + chunk B (rows 8..15)
#if USE_ASYNC_LDS
    const __bf16* gsrc = &gis[t & 1][0];
    const int gstride = LDGI_;
#else
    const __bf16* gsrc = gi + ((size_t)t * B_ + bbase) * G_;
    const int gstride = G_;
#endif
#pragma unroll
    for (int half = 0; half < 2; ++half) {
      const int r = half ? rgB : rgA;
      const int c = half ? cgB : cgA;
      const __bf16* row = gsrc + r * gstride;
      v4bf ir4 = *(const v4bf*)(row + c);
      v4bf iz4 = *(const v4bf*)(row + c + H_);
      v4bf in4 = *(const v4bf*)(row + c + 2 * H_);
      v4f  hr4 = *(const v4f*)(gh + r * LDGH_ + c);
      v4f  hz4 = *(const v4f*)(gh + r * LDGH_ + c + H_);
      v4f  hn4 = *(const v4f*)(gh + r * LDGH_ + c + 2 * H_);
      v4f  hp4 = *(const v4f*)(hs + r * H_ + c);
      v4f  hnew;
      v4bf hnb;
#pragma unroll
      for (int e = 0; e < 4; ++e) {
        const float rg = 1.f / (1.f + __expf(-((float)ir4[e] + hr4[e])));
        const float zg = 1.f / (1.f + __expf(-((float)iz4[e] + hz4[e])));
        const float e2 = __expf(-2.f * ((float)in4[e] + rg * hn4[e]));
        const float ng = (1.f - e2) / (1.f + e2);       // tanh
        const float hv = (1.f - zg) * ng + zg * hp4[e];
        hnew[e] = hv;
        hnb[e]  = (__bf16)hv;
      }
      *(v4f*) (hs  + r * H_   + c) = hnew;
      *(v4bf*)(hbf + r * LDA_ + c) = hnb;
    }
#if !USE_ASYNC_LDS
    if (t + 2 < T_) {   // fallback: pull gi[t+2] toward the WGP caches
      const char* pf = (const char*)(gi + ((size_t)(t + 2) * B_ + bbase) * G_);
      __builtin_prefetch(pf + tid * 48, 0, 1);
    }
#endif
    __syncthreads();
  }

  for (int idx = tid; idx < 16 * H_; idx += 512)
    out[(bbase + (idx >> 8)) * H_ + (idx & 255)] = hs[idx];
}

// ---------------- launcher --------------------------------------------------
extern "C" void kernel_launch(void* const* d_in, const int* in_sizes, int n_in,
                              void* d_out, int out_size, void* d_ws, size_t ws_size,
                              hipStream_t stream) {
  (void)in_sizes; (void)n_in; (void)out_size; (void)ws_size;
  const int*   seq = (const int*)  d_in[0];
  const float* emb = (const float*)d_in[1];
  const float* wih = (const float*)d_in[2];
  const float* whh = (const float*)d_in[3];
  const float* bih = (const float*)d_in[4];
  const float* bhh = (const float*)d_in[5];
  float* out = (float*)d_out;

  char* ws = (char*)d_ws;
  __bf16* wihb = (__bf16*)ws;
  __bf16* whhb = (__bf16*)(ws + (size_t)G_ * E_ * sizeof(__bf16));
  __bf16* gi   = (__bf16*)(ws + 2 * (size_t)G_ * E_ * sizeof(__bf16));

  k_cvt_w<<<(G_ * E_ + 255) / 256, 256, 0, stream>>>(wih, whh, wihb, whhb);
  k_gi  <<<(B_ * T_) / 128, 768, 0, stream>>>(seq, emb, wihb, bih, gi);
  k_scan<<<B_ / 16, 512, 0, stream>>>(whhb, bhh, gi, out);
}
